// CausalSelfAttention_19413252178704
// MI455X (gfx1250) — matrix operand order repaired and checked by reading, not compiled, not run on hardware
//
#include <hip/hip_runtime.h>

// ---------------------------------------------------------------------------
// Types for CDNA5 WMMA
// ---------------------------------------------------------------------------
typedef __bf16 v16bf __attribute__((ext_vector_type(16)));
typedef float  v8f   __attribute__((ext_vector_type(8)));

union ABFrag {
    v16bf v;
    uint4 q[2];   // 32 bytes
};

__device__ __forceinline__ unsigned short cvt_bf16(float f) {
    unsigned int u = __float_as_uint(f);
    u += 0x7FFFu + ((u >> 16) & 1u);   // round-to-nearest-even
    return (unsigned short)(u >> 16);
}

__device__ __forceinline__ v8f wmma_bf16(const ABFrag& a, const ABFrag& b, v8f c) {
    return __builtin_amdgcn_wmma_f32_16x16x32_bf16(
        /*neg_a=*/false, a.v, /*neg_b=*/false, b.v,
        /*c_mod=*/(short)0, c, /*reuse_a=*/false, /*reuse_b=*/false);
}

// Low 32 bits of a generic pointer to a __shared__ object = LDS byte offset
// (generic LDS address = {SHARED_BASE, offset[31:0]}); hardware adds LDS_BASE
// for both ds ops and async-load VDST, so this is the right coordinate.
__device__ __forceinline__ unsigned lds_off(const void* p) {
    return (unsigned)(unsigned long long)(uintptr_t)p;
}

// ---------------------------------------------------------------------------
// Conversion kernels (run once per launch; cheap vs. matrix work)
// ---------------------------------------------------------------------------
__global__ __launch_bounds__(256) void cvt_f32_bf16(const float* __restrict__ in,
                                                    unsigned short* __restrict__ out,
                                                    int n4) {
    int i = blockIdx.x * 256 + threadIdx.x;
    if (i < n4) {
        float4 f = ((const float4*)in)[i];
        ushort4 u;
        u.x = cvt_bf16(f.x); u.y = cvt_bf16(f.y);
        u.z = cvt_bf16(f.z); u.w = cvt_bf16(f.w);
        ((ushort4*)out)[i] = u;
    }
}

// in: [K,N] f32 row-major  ->  out: [N,K] bf16 row-major (coalesced on output)
__global__ __launch_bounds__(256) void transpose_cvt(const float* __restrict__ in,
                                                     unsigned short* __restrict__ out,
                                                     int K, int N) {
    int idx = blockIdx.x * 256 + threadIdx.x;
    if (idx < K * N) {
        int n = idx / K;
        int k = idx - n * K;
        out[idx] = cvt_bf16(in[(size_t)k * N + n]);
    }
}

// ---------------------------------------------------------------------------
// bf16 WMMA GEMM:  C[M,N] = A[M,K] * Bt[N,K]^T
// Block = 128 threads = 4 waves stacked along M; block tile = 128 x 64.
// Each wave owns a 32x64 C tile (2 x 4 accumulators, 8 WMMAs per k-step).
// The block-shared 64x32 weight tile is double-buffered in LDS via
// global_load_async_to_lds_b128 (ASYNCcnt), overlapping the next tile's
// fetch with the current tile's WMMAs.
// ---------------------------------------------------------------------------
template <bool OUT_BF16, bool TRANS_STORE>
__device__ __forceinline__ void store_acc(void* C, const v8f& acc,
                                          int m0, int n0, int M, int N,
                                          int half, int l16) {
#pragma unroll
    for (int g = 0; g < 8; ++g) {
        int m = m0 + g + half * 8;
        int n = n0 + l16;
        if (OUT_BF16) {
            unsigned short* o = (unsigned short*)C;
            size_t idx = TRANS_STORE ? ((size_t)n * M + m) : ((size_t)m * N + n);
            o[idx] = cvt_bf16(acc[g]);
        } else {
            ((float*)C)[(size_t)m * N + n] = acc[g];
        }
    }
}

template <bool OUT_BF16, bool TRANS_STORE>
__global__ __launch_bounds__(128) void gemm_bf16_kernel(
    const unsigned short* __restrict__ A,    // [M,K] bf16
    const unsigned short* __restrict__ Bt,   // [N,K] bf16 (pre-transposed)
    void* __restrict__ C,
    int M, int N, int K) {
    __shared__ __align__(16) unsigned short bstage[2][64 * 32];  // 2 x 4 KB

    const int tid  = threadIdx.x;
    const int lane = tid & 31;
    const int wave = tid >> 5;
    const int half = lane >> 4;
    const int l16  = lane & 15;

    const int m0 = blockIdx.y * 128 + wave * 32;
    const int n0 = blockIdx.x * 64;

    // --- async staging of Bt tile [n0..n0+63][k..k+31] (row = n, 64 B/row) ---
    const int srow   = tid >> 2;          // 0..31 (two passes -> 64 rows)
    const int schunk = (tid & 3) * 16;    // 16-byte chunk within a row
    const unsigned long long bbase = (unsigned long long)(uintptr_t)Bt;
    const unsigned ldsbuf[2] = { lds_off(&bstage[0][0]), lds_off(&bstage[1][0]) };

    auto issue_stage = [&](unsigned lbase, int k) {
#pragma unroll
        for (int r = 0; r < 2; ++r) {
            const int row = srow + r * 32;
            const unsigned long long ga =
                bbase + ((unsigned long long)(n0 + row) * (unsigned)K + (unsigned)k) * 2ull
                      + (unsigned)schunk;
            const unsigned la = lbase + (unsigned)(row * 64 + schunk);
            asm volatile("global_load_async_to_lds_b128 %0, %1, off"
                         :: "v"(la), "v"(ga) : "memory");
        }
    };

    v8f acc[2][4];
#pragma unroll
    for (int mi = 0; mi < 2; ++mi)
#pragma unroll
        for (int j = 0; j < 4; ++j) acc[mi][j] = {};

    const uint4* Aq = (const uint4*)A;
    const int ra0 = m0 + l16;
    const int ra1 = m0 + 16 + l16;

    issue_stage(ldsbuf[0], 0);   // prologue

    for (int k = 0, it = 0; k < K; k += 32, ++it) {
        __syncthreads();   // everyone done reading the buffer we're about to fill
        if (k + 32 < K) {
            issue_stage(ldsbuf[(it + 1) & 1], k + 32);
            asm volatile("s_wait_asynccnt 1" ::: "memory");  // current tile done
        } else {
            asm volatile("s_wait_asynccnt 0" ::: "memory");
        }
        __syncthreads();   // current tile visible to all waves

        const uint4* Bl = (const uint4*)&bstage[it & 1][0];

        ABFrag a0, a1;
        const int ka = k + half * 16;
        a0.q[0] = Aq[((size_t)ra0 * K + ka) >> 3];
        a0.q[1] = Aq[((size_t)ra0 * K + ka + 8) >> 3];
        a1.q[0] = Aq[((size_t)ra1 * K + ka) >> 3];
        a1.q[1] = Aq[((size_t)ra1 * K + ka + 8) >> 3];

#pragma unroll
        for (int j = 0; j < 4; ++j) {
            ABFrag b;
            const int row = j * 16 + l16;
            b.q[0] = Bl[row * 4 + half * 2];
            b.q[1] = Bl[row * 4 + half * 2 + 1];
            acc[0][j] = wmma_bf16(a0, b, acc[0][j]);
            acc[1][j] = wmma_bf16(a1, b, acc[1][j]);
        }
    }

#pragma unroll
    for (int mi = 0; mi < 2; ++mi)
#pragma unroll
        for (int j = 0; j < 4; ++j)
            store_acc<OUT_BF16, TRANS_STORE>(C, acc[mi][j],
                                             m0 + mi * 16, n0 + j * 16,
                                             M, N, half, l16);
}

// ---------------------------------------------------------------------------
// Flash-style windowed attention.
//   qk : [B*T, 2*H*DQK] bf16   (q cols [h*DQK, ...), k cols [H*DQK + h*DQK, ...))
//   vT : [H*64, B*T]     bf16  (value, transposed)
//   Y  : [B*T, 1024]     bf16  (concat output; headBase = 0 short, 512 long)
// One wave handles one (b, h, 16-query tile). 4 independent waves per block
// (no cross-wave sync: chunk counts differ per wave near the window edge).
// ---------------------------------------------------------------------------
template <int DQK, int WIN>
__global__ __launch_bounds__(128) void attn_kernel(
    const unsigned short* __restrict__ qk,
    const unsigned short* __restrict__ vT,
    unsigned short* __restrict__ Y,
    int headBase) {
    constexpr int T = 2048;
    constexpr int H = 8;
    constexpr int LD = 2 * H * DQK;
    constexpr float SCALE = (DQK == 32) ? 0.17677669529663687f : 0.08838834764831843f;

    __shared__ __align__(16) unsigned short pbuf[4][16 * 32];  // per-wave P tile

    const int lane = threadIdx.x & 31;
    const int wv   = threadIdx.x >> 5;
    const int half = lane >> 4;
    const int l16  = lane & 15;

    const int gw = blockIdx.x * 4 + wv;   // global wave id: [B*H*(T/16))
    const int qt = gw & 127;              // query tile
    const int h  = (gw >> 7) & 7;         // head
    const int bb = gw >> 10;              // batch
    const int q0 = qt * 16;

    const uint4* Q  = (const uint4*)qk;
    const uint4* Vq = (const uint4*)vT;

    v8f o0 = {}, o1 = {}, o2 = {}, o3 = {};
    float mrun[8], lrun[8];
#pragma unroll
    for (int g = 0; g < 8; ++g) { mrun[g] = -1e30f; lrun[g] = 0.f; }

    int kstart = q0 - WIN + 1;
    if (kstart < 0) kstart = 0;
    const int kc0 = kstart & ~31;

    for (int kc = kc0; kc <= q0 + 15; kc += 32) {
        // ---- scores S = Q * K^T  (16 x 32 chunk as two 16x16 tiles) ----
        v8f s0 = {}, s1 = {};
        const int qrow = q0 + l16;
        const int key0 = kc + l16;        // <= 2047 (T multiple of 32)
        const int key1 = kc + 16 + l16;
#pragma unroll
        for (int kk = 0; kk < DQK; kk += 32) {
            ABFrag qa, kb0, kb1;
            const int qcol = h * DQK + kk + half * 16;
            const size_t qb = (size_t)(bb * T + qrow) * LD + qcol;
            qa.q[0] = Q[qb >> 3];
            qa.q[1] = Q[(qb + 8) >> 3];

            const int kcol = H * DQK + h * DQK + kk + half * 16;
            const size_t k0b = (size_t)(bb * T + key0) * LD + kcol;
            const size_t k1b = (size_t)(bb * T + key1) * LD + kcol;
            kb0.q[0] = Q[k0b >> 3]; kb0.q[1] = Q[(k0b + 8) >> 3];
            kb1.q[0] = Q[k1b >> 3]; kb1.q[1] = Q[(k1b + 8) >> 3];

            s0 = wmma_bf16(qa, kb0, s0);
            s1 = wmma_bf16(qa, kb1, s1);
        }

        // ---- mask + online softmax (rows live in 16-lane halves) ----
        float pr0[8], pr1[8], alpha[8];
#pragma unroll
        for (int g = 0; g < 8; ++g) {
            const int tq  = q0 + g + half * 8;
            const int d0  = tq - (kc + l16);
            const int d1  = tq - (kc + 16 + l16);
            const bool v0 = (d0 >= 0) && (d0 < WIN);
            const bool v1 = (d1 >= 0) && (d1 < WIN);
            const float x0 = v0 ? s0[g] * SCALE : -1e30f;
            const float x1 = v1 ? s1[g] * SCALE : -1e30f;

            float mx = fmaxf(x0, x1);
            mx = fmaxf(mx, __shfl_xor(mx, 1));
            mx = fmaxf(mx, __shfl_xor(mx, 2));
            mx = fmaxf(mx, __shfl_xor(mx, 4));
            mx = fmaxf(mx, __shfl_xor(mx, 8));

            const float mnew = fmaxf(mrun[g], mx);
            const float al   = __expf(mrun[g] - mnew);
            mrun[g] = mnew;

            const float p0 = v0 ? __expf(x0 - mnew) : 0.f;
            const float p1 = v1 ? __expf(x1 - mnew) : 0.f;
            float rs = p0 + p1;
            rs += __shfl_xor(rs, 1);
            rs += __shfl_xor(rs, 2);
            rs += __shfl_xor(rs, 4);
            rs += __shfl_xor(rs, 8);

            lrun[g] = lrun[g] * al + rs;
            alpha[g] = al;
            pr0[g] = p0;
            pr1[g] = p1;
        }
#pragma unroll
        for (int g = 0; g < 8; ++g) {
            o0[g] *= alpha[g]; o1[g] *= alpha[g];
            o2[g] *= alpha[g]; o3[g] *= alpha[g];
        }

        // ---- P: C-layout regs -> LDS -> A-fragment (per-wave, fenced) ----
        unsigned short* pb = &pbuf[wv][0];
#pragma unroll
        for (int g = 0; g < 8; ++g) {
            const int r = g + half * 8;
            pb[r * 32 + l16]      = cvt_bf16(pr0[g]);
            pb[r * 32 + 16 + l16] = cvt_bf16(pr1[g]);
        }
        asm volatile("s_wait_dscnt 0" ::: "memory");   // intra-wave LDS fence

        ABFrag pa;
        const uint4* Pq = (const uint4*)pb;
        pa.q[0] = Pq[((size_t)l16 * 32 + half * 16) >> 3];
        pa.q[1] = Pq[((size_t)l16 * 32 + half * 16 + 8) >> 3];

        // ---- O += P * V  (16 x 64 via 4 WMMAs) ----
#pragma unroll
        for (int j = 0; j < 4; ++j) {
            ABFrag vb;
            const size_t vbase =
                (size_t)(h * 64 + j * 16 + l16) * (2 * T) + (size_t)bb * T + kc + half * 16;
            vb.q[0] = Vq[vbase >> 3];
            vb.q[1] = Vq[(vbase + 8) >> 3];
            if (j == 0) o0 = wmma_bf16(pa, vb, o0);
            if (j == 1) o1 = wmma_bf16(pa, vb, o1);
            if (j == 2) o2 = wmma_bf16(pa, vb, o2);
            if (j == 3) o3 = wmma_bf16(pa, vb, o3);
        }
    }

    // ---- normalize + write y (bf16, concat layout) ----
#pragma unroll
    for (int g = 0; g < 8; ++g) {
        const float inv = 1.0f / lrun[g];   // diagonal always in-window -> lrun > 0
        const int t = q0 + g + half * 8;
        unsigned short* yrow = Y + (size_t)(bb * T + t) * 1024 + headBase + h * 64;
        yrow[ 0 + l16] = cvt_bf16(o0[g] * inv);
        yrow[16 + l16] = cvt_bf16(o1[g] * inv);
        yrow[32 + l16] = cvt_bf16(o2[g] * inv);
        yrow[48 + l16] = cvt_bf16(o3[g] * inv);
    }
}

// ---------------------------------------------------------------------------
// Host orchestration
// ---------------------------------------------------------------------------
extern "C" void kernel_launch(void* const* d_in, const int* in_sizes, int n_in,
                              void* d_out, int out_size, void* d_ws, size_t ws_size,
                              hipStream_t stream) {
    (void)in_sizes; (void)n_in; (void)out_size; (void)ws_size;

    constexpr int B = 2, T = 2048, C = 1024;
    constexpr int M = B * T;                  // 4096 rows
    constexpr int NQK_S = 512, NV = 512, NQK_L = 2048;

    const float* x     = (const float*)d_in[0];
    const float* WqkS  = (const float*)d_in[1];
    const float* WvS   = (const float*)d_in[2];
    const float* WqkL  = (const float*)d_in[3];
    const float* WvL   = (const float*)d_in[4];
    const float* Wproj = (const float*)d_in[5];
    float* out = (float*)d_out;

    char* ws = (char*)d_ws;
    auto alloc = [&](size_t bytes) -> unsigned short* {
        char* p = ws;
        ws += (bytes + 255) & ~(size_t)255;
        return (unsigned short*)p;
    };

    unsigned short* xb      = alloc((size_t)M * C * 2);        // x bf16
    unsigned short* WqkS_t  = alloc((size_t)NQK_S * C * 2);    // [N,K]
    unsigned short* WvS_t   = alloc((size_t)NV * C * 2);
    unsigned short* WqkL_t  = alloc((size_t)NQK_L * C * 2);
    unsigned short* WvL_t   = alloc((size_t)NV * C * 2);
    unsigned short* Wproj_t = alloc((size_t)C * C * 2);
    unsigned short* qk_s    = alloc((size_t)M * NQK_S * 2);    // [M, 512]
    unsigned short* vT_s    = alloc((size_t)NV * M * 2);       // [512, M]
    unsigned short* qk_l    = alloc((size_t)M * NQK_L * 2);    // [M, 2048]
    unsigned short* vT_l    = alloc((size_t)NV * M * 2);       // [512, M]
    unsigned short* yb      = alloc((size_t)M * C * 2);        // concat y bf16

    // 1) conversions
    {
        int n4 = (M * C) / 4;
        cvt_f32_bf16<<<(n4 + 255) / 256, 256, 0, stream>>>(x, xb, n4);
    }
    auto tr = [&](const float* in, unsigned short* o, int K, int N) {
        int tot = K * N;
        transpose_cvt<<<(tot + 255) / 256, 256, 0, stream>>>(in, o, K, N);
    };
    tr(WqkS,  WqkS_t,  C, NQK_S);
    tr(WvS,   WvS_t,   C, NV);
    tr(WqkL,  WqkL_t,  C, NQK_L);
    tr(WvL,   WvL_t,   C, NV);
    tr(Wproj, Wproj_t, C, C);

    // 2) input projections (WMMA bf16, async-staged weights)
    gemm_bf16_kernel<true, false><<<dim3(NQK_S / 64, M / 128), 128, 0, stream>>>(
        xb, WqkS_t, qk_s, M, NQK_S, C);
    gemm_bf16_kernel<true, true ><<<dim3(NV / 64, M / 128), 128, 0, stream>>>(
        xb, WvS_t, vT_s, M, NV, C);
    gemm_bf16_kernel<true, false><<<dim3(NQK_L / 64, M / 128), 128, 0, stream>>>(
        xb, WqkL_t, qk_l, M, NQK_L, C);
    gemm_bf16_kernel<true, true ><<<dim3(NV / 64, M / 128), 128, 0, stream>>>(
        xb, WvL_t, vT_l, M, NV, C);

    // 3) windowed flash attention; 2048 wave-tiles per branch, 4 waves/block
    attn_kernel<32, 256><<<512, 128, 0, stream>>>(qk_s, vT_s, yb, /*headBase=*/0);
    attn_kernel<128, 1024><<<512, 128, 0, stream>>>(qk_l, vT_l, yb, /*headBase=*/512);

    // 4) output projection -> f32
    gemm_bf16_kernel<false, false><<<dim3(C / 64, M / 128), 128, 0, stream>>>(
        yb, Wproj_t, out, M, C, C);
}
